// AttentionBlock_88210038325306
// MI455X (gfx1250) — matrix operand order repaired and checked by reading, not compiled, not run on hardware
//
#include <hip/hip_runtime.h>
#include <stdint.h>

// Problem dims (fixed by the reference)
#define B_ 8
#define T_ 2048
#define C_ 512
#define M_ (B_ * T_)          // 16384 rows
#define INV_SQRT_K 0.04419417382415922f  // 1/sqrt(512)

typedef __attribute__((ext_vector_type(16))) __bf16       v16bf;
typedef __attribute__((ext_vector_type(8)))  float        v8f;
typedef __attribute__((ext_vector_type(4)))  unsigned int u32x4;

union BFrag {
  u32x4          u4[2];
  unsigned short us[16];
  v16bf          v;
};

__device__ __forceinline__ unsigned short f2bf(float f) {
  union { float f; unsigned u; } c; c.f = f;
  unsigned u = c.u;
  u += 0x7fffu + ((u >> 16) & 1u);   // round-to-nearest-even
  return (unsigned short)(u >> 16);
}

__device__ __forceinline__ v8f wmma_bf16(v16bf a, v16bf b, v8f c) {
  return __builtin_amdgcn_wmma_f32_16x16x32_bf16(
      /*neg_a=*/false, a, /*neg_b=*/false, b,
      /*c_mod=*/(short)0, c, /*reuse_a=*/false, /*reuse_b=*/false);
}

// ---------------------------------------------------------------------------
// Kernel 0a: input f32 -> bf16 scratch, and write concat first half of output
// ---------------------------------------------------------------------------
__global__ void __launch_bounds__(256) k_prep_x(const float* __restrict__ x,
                                               float* __restrict__ out,
                                               unsigned short* __restrict__ xbf) {
  int idx = blockIdx.x * 256 + threadIdx.x;     // < M_*C_
  float v = x[idx];
  int row = idx >> 9, c = idx & 511;
  out[(size_t)row * 1024 + c] = v;              // out[..., 0:512] = input
  xbf[idx] = f2bf(v);
}

// ---------------------------------------------------------------------------
// Kernel 0b: weights [K][C] f32 -> transposed bf16 Wt[C][K] (q,k,v packed)
// ---------------------------------------------------------------------------
__global__ void __launch_bounds__(256) k_prep_w(const float* __restrict__ Wq,
                                               const float* __restrict__ Wk,
                                               const float* __restrict__ Wv,
                                               unsigned short* __restrict__ wT) {
  int idx = blockIdx.x * 256 + threadIdx.x;     // < 3*512*512
  int which = idx >> 18;
  int rem = idx & 262143;
  int n = rem >> 9, c = rem & 511;
  const float* W = (which == 0) ? Wq : (which == 1) ? Wk : Wv;
  wT[(which << 18) + (c << 9) + n] = f2bf(W[(n << 9) + c]);
}

// ---------------------------------------------------------------------------
// Kernel 1: QKV projection GEMM.  D = Xbf[16384,512] * WtT + bias.
// grid (M/128, 512/128, 3); 8 waves per block arranged 4(M) x 2(N),
// wave tile 32(M) x 64(N) = 2x4 WMMA tiles.
// Q -> row-major bf16, V -> row-major bf16,
// K -> transposed [b][k][t] bf16 via LDS-staged tile for coalesced stores.
// ---------------------------------------------------------------------------
#define TSTR 136   // padded LDS row stride (ushorts) to avoid bank conflicts
__global__ void __launch_bounds__(256) k_qkv(const unsigned short* __restrict__ xbf,
                                             const unsigned short* __restrict__ wT,
                                             const float* __restrict__ bq,
                                             const float* __restrict__ bk,
                                             const float* __restrict__ bv,
                                             unsigned short* __restrict__ qbf,
                                             unsigned short* __restrict__ ktbf,
                                             unsigned short* __restrict__ vbf) {
  __shared__ unsigned short tlds[128 * TSTR];   // transpose staging (K only)
  const int which = blockIdx.z;
  const unsigned short* w = wT + (which << 18);
  const int m0 = blockIdx.x * 128, n0 = blockIdx.y * 128;
  const int wave = threadIdx.x >> 5, lane = threadIdx.x & 31;
  const int mw = m0 + (wave & 3) * 32;  // 2 M tiles
  const int nw = n0 + (wave >> 2) * 64; // 4 N tiles
  const int mA = lane & 15, kgrp = (lane >> 4) * 8;

  v8f acc[2][4] = {};
#pragma unroll 4
  for (int k = 0; k < 512; k += 32) {
    if (k + 32 < 512)  // prefetch next k-step's B rows (global_prefetch_b8)
      __builtin_prefetch(w + (k + 32 + lane) * 512 + nw, 0, 1);
    BFrag a[2];
#pragma unroll
    for (int mt = 0; mt < 2; ++mt) {
      const unsigned short* ar = xbf + (mw + mt * 16 + mA) * 512 + k;
      a[mt].u4[0] = *(const u32x4*)(ar + kgrp);
      a[mt].u4[1] = *(const u32x4*)(ar + 16 + kgrp);
    }
    BFrag b[4];
#pragma unroll
    for (int nt = 0; nt < 4; ++nt) {
      const unsigned short* br = w + (k + kgrp) * 512 + nw + nt * 16 + mA;
#pragma unroll
      for (int s = 0; s < 8; ++s) {
        b[nt].us[s] = br[s * 512];
        b[nt].us[8 + s] = br[(16 + s) * 512];
      }
    }
#pragma unroll
    for (int mt = 0; mt < 2; ++mt)
#pragma unroll
      for (int nt = 0; nt < 4; ++nt)
        acc[mt][nt] = wmma_bf16(a[mt].v, b[nt].v, acc[mt][nt]);
  }

  const float* bias = (which == 0) ? bq : (which == 1) ? bk : bv;
  const int rl = lane >> 4, nc = lane & 15;
  if (which != 1) {
    unsigned short* dst = (which == 0) ? qbf : vbf;
#pragma unroll
    for (int mt = 0; mt < 2; ++mt)
#pragma unroll
      for (int nt = 0; nt < 4; ++nt) {
        int ncol = nw + nt * 16 + nc;
        float bb = bias[ncol];
#pragma unroll
        for (int r = 0; r < 8; ++r) {
          int m = mw + mt * 16 + r + 8 * rl;
          dst[m * 512 + ncol] = f2bf(acc[mt][nt][r] + bb);
        }
      }
  } else {
    // Stage transposed tile [ncol][t] in LDS, then store coalesced rows.
#pragma unroll
    for (int mt = 0; mt < 2; ++mt)
#pragma unroll
      for (int nt = 0; nt < 4; ++nt) {
        int ncol_l = (nw - n0) + nt * 16 + nc;
        float bb = bias[nw + nt * 16 + nc];
#pragma unroll
        for (int r = 0; r < 8; ++r) {
          int m_l = (mw - m0) + mt * 16 + r + 8 * rl;
          tlds[ncol_l * TSTR + m_l] = f2bf(acc[mt][nt][r] + bb);
        }
      }
    __syncthreads();
    const int b = m0 >> 11, t0 = m0 & 2047;
    const int tid = threadIdx.x;
#pragma unroll
    for (int c = 0; c < 8; ++c) {
      int chunk = c * 256 + tid;        // 0..2047, lanes -> contiguous 16B
      int rowl = chunk >> 4;            // 0..127 (ncol)
      int offl = (chunk & 15) * 8;      // 0..120 (t)
      u32x4 vv = *(const u32x4*)(tlds + rowl * TSTR + offl);
      *(u32x4*)(ktbf + (size_t)(b * 512 + n0 + rowl) * 2048 + t0 + offl) = vv;
    }
  }
}

// ---------------------------------------------------------------------------
// Kernel 2: per-column softmax stats (softmax over query axis j, j >= i).
// grid (T/128, B); wave w owns 16 columns i in [i0+16w, i0+16w+16),
// streams j-tiles via WMMA, online max/sum per lane, shfl-xor(16) merge.
// ---------------------------------------------------------------------------
__global__ void __launch_bounds__(256) k_stats(const unsigned short* __restrict__ qbf,
                                               const unsigned short* __restrict__ ktbf,
                                               float* __restrict__ colmax,
                                               float* __restrict__ rdenom) {
  const int b = blockIdx.y;
  const int wave = threadIdx.x >> 5, lane = threadIdx.x & 31;
  const int iw = blockIdx.x * 128 + wave * 16;       // this wave's 16 columns
  const unsigned short* qb = qbf + b * T_ * 512;
  const unsigned short* ktb = ktbf + b * 512 * 2048;
  const int mA = lane & 15, kgrp = (lane >> 4) * 8, rl = lane >> 4;
  const int ig = iw + (lane & 15);                   // column this lane tracks
  const float NEG = -__builtin_inff();

  float mrun = NEG, srun = 0.f;
  for (int j0 = iw; j0 < T_; j0 += 16) {
    if (j0 + 16 < T_)  // prefetch next j-tile's Q rows
      __builtin_prefetch(qb + (j0 + 16 + mA) * 512 + (lane >> 4) * 256, 0, 1);
    v8f s = {};
#pragma unroll 4
    for (int k = 0; k < 512; k += 32) {
      BFrag a, bf;
      const unsigned short* ar = qb + (j0 + mA) * 512 + k;
      a.u4[0] = *(const u32x4*)(ar + kgrp);
      a.u4[1] = *(const u32x4*)(ar + 16 + kgrp);
      const unsigned short* br = ktb + (k + kgrp) * 2048 + iw + mA;
#pragma unroll
      for (int s = 0; s < 8; ++s) {
        bf.us[s] = br[s * 2048];
        bf.us[8 + s] = br[(16 + s) * 2048];
      }
      s = wmma_bf16(a.v, bf.v, s);
    }
    float vals[8], tmax = NEG;
#pragma unroll
    for (int r = 0; r < 8; ++r) {
      int j = j0 + r + 8 * rl;
      float v = (ig <= j) ? s[r] * INV_SQRT_K : NEG;
      vals[r] = v;
      tmax = fmaxf(tmax, v);
    }
    float mnew = fmaxf(mrun, tmax);
    if (mnew > -1e38f) {
      srun *= (mrun > -1e38f) ? __expf(mrun - mnew) : 0.f;
#pragma unroll
      for (int r = 0; r < 8; ++r)
        if (vals[r] > -1e38f) srun += __expf(vals[r] - mnew);
      mrun = mnew;
    }
  }
  // merge lane L with L^16 (same column, other j-half)
  float mo = __shfl_xor(mrun, 16, 32);
  float so = __shfl_xor(srun, 16, 32);
  float mm = fmaxf(mrun, mo);
  float ss = srun * __expf(mrun - mm) + so * __expf(mo - mm);
  if (lane < 16) {
    colmax[b * T_ + ig] = mm;
    rdenom[b * T_ + ig] = 1.f / ss;
  }
}

// ---------------------------------------------------------------------------
// Kernel 3: read = P * V, fused logit recompute + normalized exp.
// grid (T/64, B); 8 waves: pair p=wave&3 owns 16 j-rows, half h=wave>>2 is
// BOTH the K-half for computing S (split-K, no duplicated logit work) AND the
// v-half for P*V.  Partial f32 S tiles are exchanged through LDS with a
// workgroup barrier; P goes through a padded per-wave LDS tile to become a
// bf16 A-fragment (i sits on lane%16, so colmax/rdenom are per-lane scalars).
// ---------------------------------------------------------------------------
#define PSTR 40    // padded P-tile row stride (ushorts)
__global__ void __launch_bounds__(256) k_attn(const unsigned short* __restrict__ qbf,
                                              const unsigned short* __restrict__ ktbf,
                                              const unsigned short* __restrict__ vbf,
                                              const float* __restrict__ colmax,
                                              const float* __restrict__ rdenom,
                                              float* __restrict__ out) {
  __shared__ float spart[4][2][16 * 32];          // partial S exchange (16KB)
  __shared__ unsigned short ptile[8][16 * PSTR];  // per-wave bf16 P tiles
  const int b = blockIdx.y;
  const int j0 = blockIdx.x * 64;
  const int wave = threadIdx.x >> 5, lane = threadIdx.x & 31;
  const int p = wave & 3, h = wave >> 2;
  const int jw = j0 + p * 16;
  const int vbase = h * 256;
  const int kbeg = h * 256, kend = kbeg + 256;    // this wave's K-half for S
  const unsigned short* qb = qbf + b * T_ * 512;
  const unsigned short* ktb = ktbf + b * 512 * 2048;
  const unsigned short* vb = vbf + b * T_ * 512;
  const float* cmb = colmax + b * T_;
  const float* rdb = rdenom + b * T_;
  const int mA = lane & 15, kgrp = (lane >> 4) * 8, rl = lane >> 4;
  unsigned short* pl = ptile[wave];
  float* spw = &spart[p][h][0];
  const float* spr = &spart[p][1 - h][0];

  v8f acc[16] = {};
  const int ilim = j0 + 64;                  // causal: only i <= j contribute
  for (int i0 = 0; i0 < ilim; i0 += 32) {
    // ---- partial S tiles: 16(j) x 32(i) over this wave's K-half ----
    v8f s0 = {}, s1 = {};
#pragma unroll 4
    for (int k = kbeg; k < kend; k += 32) {
      BFrag a, b0, b1;
      const unsigned short* ar = qb + (jw + mA) * 512 + k;
      a.u4[0] = *(const u32x4*)(ar + kgrp);
      a.u4[1] = *(const u32x4*)(ar + 16 + kgrp);
      const unsigned short* br = ktb + (k + kgrp) * 2048 + i0 + mA;
#pragma unroll
      for (int s = 0; s < 8; ++s) {
        b0.us[s] = br[s * 2048];
        b0.us[8 + s] = br[(16 + s) * 2048];
        b1.us[s] = br[s * 2048 + 16];
        b1.us[8 + s] = br[(16 + s) * 2048 + 16];
      }
      s0 = wmma_bf16(a.v, b0.v, s0);
      s1 = wmma_bf16(a.v, b1.v, s1);
    }
    // ---- exchange partials with the partner wave ----
#pragma unroll
    for (int r = 0; r < 8; ++r) {
      int jl = r + 8 * rl;
      spw[jl * 32 + (lane & 15)] = s0[r];
      spw[jl * 32 + 16 + (lane & 15)] = s1[r];
    }
    __syncthreads();
#pragma unroll
    for (int r = 0; r < 8; ++r) {
      int jl = r + 8 * rl;
      s0[r] += spr[jl * 32 + (lane & 15)];
      s1[r] += spr[jl * 32 + 16 + (lane & 15)];
    }
    // ---- masked, normalized exp in D layout; spill bf16 P tile to LDS ----
    const int igl0 = i0 + (lane & 15);
    const int igl1 = igl0 + 16;
    const float cm0 = cmb[igl0], rd0 = rdb[igl0];
    const float cm1 = cmb[igl1], rd1 = rdb[igl1];
#pragma unroll
    for (int r = 0; r < 8; ++r) {
      int jl = r + 8 * rl;
      int jg = jw + jl;
      float p0 = (igl0 <= jg) ? __expf(s0[r] * INV_SQRT_K - cm0) * rd0 : 0.f;
      float p1 = (igl1 <= jg) ? __expf(s1[r] * INV_SQRT_K - cm1) * rd1 : 0.f;
      pl[jl * PSTR + (lane & 15)] = f2bf(p0);
      pl[jl * PSTR + 16 + (lane & 15)] = f2bf(p1);
    }
    asm volatile("s_wait_dscnt 0" ::: "memory");  // same-wave LDS RAW
    // ---- reload P as a 16x32 bf16 A-fragment ----
    BFrag pa;
    pa.u4[0] = *(const u32x4*)(pl + mA * PSTR + kgrp);
    pa.u4[1] = *(const u32x4*)(pl + mA * PSTR + 16 + kgrp);
    // ---- acc += P * V  (V rows i0..i0+31, 16 n-tiles on this v-half) ----
#pragma unroll
    for (int nt = 0; nt < 16; ++nt) {
      BFrag bv_;
      const unsigned short* vr = vb + (i0 + kgrp) * 512 + vbase + nt * 16 + mA;
#pragma unroll
      for (int s = 0; s < 8; ++s) {
        bv_.us[s] = vr[s * 512];
        bv_.us[8 + s] = vr[(16 + s) * 512];
      }
      acc[nt] = wmma_bf16(pa.v, bv_.v, acc[nt]);
    }
    __syncthreads();   // protect spart WAR before next iteration
  }
  // ---- epilogue: out[b, j, 512 + v]  (coalesced across lanes) ----
#pragma unroll
  for (int nt = 0; nt < 16; ++nt) {
    int col = vbase + nt * 16 + (lane & 15);
#pragma unroll
    for (int r = 0; r < 8; ++r) {
      int j = jw + r + 8 * rl;
      out[(size_t)(b * T_ + j) * 1024 + 512 + col] = acc[nt][r];
    }
  }
}

// ---------------------------------------------------------------------------
extern "C" void kernel_launch(void* const* d_in, const int* in_sizes, int n_in,
                              void* d_out, int out_size, void* d_ws, size_t ws_size,
                              hipStream_t stream) {
  (void)in_sizes; (void)n_in; (void)out_size; (void)ws_size;
  const float* x  = (const float*)d_in[0];
  const float* Wq = (const float*)d_in[1];
  const float* bq = (const float*)d_in[2];
  const float* Wk = (const float*)d_in[3];
  const float* bk = (const float*)d_in[4];
  const float* Wv = (const float*)d_in[5];
  const float* bv = (const float*)d_in[6];
  float* out = (float*)d_out;

  char* ws = (char*)d_ws;
  size_t o = 0;
  unsigned short* xbf  = (unsigned short*)(ws + o); o += (size_t)M_ * 512 * 2;
  unsigned short* qbf  = (unsigned short*)(ws + o); o += (size_t)M_ * 512 * 2;
  unsigned short* ktbf = (unsigned short*)(ws + o); o += (size_t)M_ * 512 * 2;
  unsigned short* vbf  = (unsigned short*)(ws + o); o += (size_t)M_ * 512 * 2;
  unsigned short* wT   = (unsigned short*)(ws + o); o += (size_t)3 * 512 * 512 * 2;
  float* cmax = (float*)(ws + o); o += (size_t)M_ * 4;
  float* rden = (float*)(ws + o); o += (size_t)M_ * 4;

  k_prep_x<<<dim3(M_ * 512 / 256), dim3(256), 0, stream>>>(x, out, xbf);
  k_prep_w<<<dim3(3 * 512 * 512 / 256), dim3(256), 0, stream>>>(Wq, Wk, Wv, wT);
  k_qkv<<<dim3(M_ / 128, 4, 3), dim3(256), 0, stream>>>(xbf, wT, bq, bk, bv,
                                                        qbf, ktbf, vbf);
  k_stats<<<dim3(T_ / 128, B_), dim3(256), 0, stream>>>(qbf, ktbf, cmax, rden);
  k_attn<<<dim3(T_ / 64, B_), dim3(256), 0, stream>>>(qbf, ktbf, vbf, cmax, rden, out);
}
